// SSMEvaluator_25151328485660
// MI455X (gfx1250) — matrix operand order repaired and checked by reading, not compiled, not run on hardware
//
#include <hip/hip_runtime.h>

// ---------------------------------------------------------------------------
// CDNA5 (gfx1250) persistent-scan SSM evaluator.
//  * Row-independent recurrence -> each block owns 16 rows for all 512 steps.
//  * All GEMMs via v_wmma_f32_16x16x32_bf16 (wave32).
//  * Weights pre-packed into WMMA B-tile layout in d_ws (bf16), L2-resident,
//    streamed with global_load_b128 + global_prefetch_b8 for the next tile.
//  * Activations staged in LDS bf16, A tiles loaded in contiguous k order (k = 16h + s), as packed B.
// ---------------------------------------------------------------------------

typedef __attribute__((ext_vector_type(16))) __bf16 v16bf;
typedef __attribute__((ext_vector_type(8)))  float  v8f;

union V16U { v16bf v; uint4 q[2]; };

// Native bf16 convert (backend emits hw cvt; replaces 5-op manual RNE twiddle)
__device__ __forceinline__ unsigned short f2bf(float f) {
  return __builtin_bit_cast(unsigned short, (__bf16)f);
}
__device__ __forceinline__ float sigmoidf_(float x) {
  return __builtin_amdgcn_rcpf(1.0f + __expf(-x));
}
__device__ __forceinline__ float softplusf_(float x) {
  return fmaxf(x, 0.0f) + __logf(1.0f + __expf(-fabsf(x)));
}

// --------- packed-weight byte offsets inside d_ws (bf16 tile-packed) --------
#define OFF_WG1   0u        // 160x256 : 5 kt x 16 nt
#define OFF_WG2   81920u    // 256x128 : 8 kt x  8 nt
#define OFF_WP1   147456u   // 160x256
#define OFF_WP2   229376u   // 256x128
#define OFF_WLIN  294912u   // 128x128 : 4 kt x 8 nt
#define OFF_WSIG  327680u   // 128x128
#define OFF_WH0   360448u   // 128x256 : 4 kt x 16 nt
#define OFF_WO0   425984u   // 256x256 : 8 kt x 16 nt
#define OFF_WH1   557056u
#define OFF_WO1   622592u
#define OFF_WE1   753664u   // 128x256
#define OFF_WE2   819200u   // 256x256
#define OFF_WE3   950272u   // 256x32 (28 padded to 32) : 8 kt x 2 nt
#define WS_TOTAL  966656u

// ---------------------------------------------------------------------------
// Pack fp32 KxN row-major weight into bf16 WMMA-B tiles.
// Tile (kt,nt) occupies 1024B at tile=(nt*numKT+kt); lane owns bytes lane*32..
// B layout (contiguous k order, same as wgemm's A reads): lane = n0 + (lane&15); lanes 0-15 hold K=k0..k0+15
// (pairs per dword), lanes 16-31 hold K=k0+16..k0+31.
// ---------------------------------------------------------------------------
__global__ void pack_w(const float* __restrict__ W, unsigned int* __restrict__ dst,
                       int Ksrc, int Nsrc, int numKT) {
  const int tile = blockIdx.x;
  const int nt = tile / numKT;
  const int kt = tile - nt * numKT;
  const int lane = threadIdx.x;
  const int n  = nt * 16 + (lane & 15);
  const int kb = kt * 32 + (lane >> 4) * 16;
  unsigned int* o = dst + tile * 256 + lane * 8;
#pragma unroll
  for (int v = 0; v < 8; ++v) {
    int k = kb + 2 * v;
    float f0 = (n < Nsrc && k     < Ksrc) ? W[k * Nsrc + n]       : 0.0f;
    float f1 = (n < Nsrc && k + 1 < Ksrc) ? W[(k + 1) * Nsrc + n] : 0.0f;
    o[v] = (unsigned int)f2bf(f0) | ((unsigned int)f2bf(f1) << 16);
  }
}

// ---------------------------------------------------------------------------
// One 16x16 output n-tile: acc(bias) += A(16xK from LDS bf16, stride 256)
//                                      x B(Kx16 packed tiles from d_ws).
// A layout (matches packed B): lane m=lane&15; lanes<16 read K
// [k0..k0+15]; lanes>=16 read [k0+16..k0+31].
// ---------------------------------------------------------------------------
__device__ __forceinline__ v8f wgemm(const unsigned short* aLDS, int acol0, int nkt,
                                     const unsigned char* btile, int lane, float bias) {
  v8f acc = {bias, bias, bias, bias, bias, bias, bias, bias};
  const int mlo = lane & 15;
  const int hiq = lane >> 4;
  const unsigned short* arow = aLDS + mlo * 256 + acol0 + hiq * 16;
  const uint4* bp = (const uint4*)btile;
  const int bl = lane * 2;
  // next n-tile's k-tile block is contiguous after this one: warm L2->L0 path
  __builtin_prefetch(btile + (unsigned)nkt * 1024u + (unsigned)(lane * 32), 0, 1);
  for (int kt = 0; kt < nkt; ++kt) {
    V16U A, B;
    A.q[0] = *(const uint4*)(arow + kt * 32);
    A.q[1] = *(const uint4*)(arow + kt * 32 + 8);
    B.q[0] = bp[kt * 64 + bl];
    B.q[1] = bp[kt * 64 + bl + 1];
    acc = __builtin_amdgcn_wmma_f32_16x16x32_bf16(false, A.v, false, B.v,
                                                  (short)0, acc, false, false);
  }
  return acc;
}

// output offsets (floats) in d_out: x_s | z | x_lp | z_lp | fractions
#define O_ZS   11010048u
#define O_XLP  78118912u
#define O_ZLP  89128960u
#define O_FR   89653248u

__global__ __launch_bounds__(128, 1) void ssm_scan(
    const float* __restrict__ input, const float* __restrict__ z0,
    const float* __restrict__ eps,   const float* __restrict__ u,
    const float* __restrict__ bg1, const float* __restrict__ bg2,
    const float* __restrict__ bp1, const float* __restrict__ bp2,
    const float* __restrict__ blin, const float* __restrict__ bsig,
    const float* __restrict__ bh0, const float* __restrict__ bo0,
    const float* __restrict__ bh1, const float* __restrict__ bo1,
    const float* __restrict__ be1, const float* __restrict__ be2,
    const float* __restrict__ be3,
    const unsigned char* __restrict__ wpk, float* __restrict__ out) {

  __shared__ __align__(16) unsigned short abuf[16 * 256];  // bf16 A staging
  __shared__ __align__(16) unsigned short bbuf[16 * 256];  // bf16 A staging (pong)
  __shared__ __align__(16) float xbuf[16 * 128];           // z state / x (f32)
  __shared__ __align__(16) float pbuf[16 * 128];           // prop
  __shared__ __align__(16) float gbuf[16 * 128];           // gate -> z_loc
  __shared__ __align__(16) float mbuf[16 * 128];           // IAF m
  __shared__ float lpbuf[16];

  const int tid  = threadIdx.x;
  const int lane = tid & 31;
  const int wid  = tid >> 5;          // 4 waves split the n-tile loops
  const int mlo  = lane & 15;
  const int hiq  = lane >> 4;
  const int rbase = blockIdx.x << 4;  // 16 rows per block

  float* outXS  = out;
  float* outZ   = out + O_ZS;
  float* outXLP = out + O_XLP;
  float* outZLP = out + O_ZLP;
  float* outFR  = out + O_FR;

  for (int idx = tid; idx < 16 * 128; idx += 128) xbuf[idx] = z0[rbase * 128 + idx];
  __syncthreads();

  for (int t = 0; t < 512; ++t) {
    if (tid < 16) lpbuf[tid] = 0.0f;
    float lpacc[8];
#pragma unroll
    for (int g = 0; g < 8; ++g) lpacc[g] = 0.0f;

    // ---- h = [xin | z_prev] -> abuf (bf16, stride 256, cols 0..159) ----
    for (int idx = tid; idx < 16 * 32; idx += 128) {
      int m = idx >> 5, d = idx & 31;
      abuf[m * 256 + d] = f2bf(input[(t * 8 + ((rbase + m) & 7)) * 32 + d]);
    }
    for (int idx = tid; idx < 16 * 128; idx += 128) {
      int m = idx >> 7, n = idx & 127;
      abuf[m * 256 + 32 + n] = f2bf(xbuf[idx]);
    }
    __syncthreads();

    // ---- g1 = relu(h@wg1+bg1) -> bbuf ----
    for (int nt = wid; nt < 16; nt += 4) {
      v8f acc = wgemm(abuf, 0, 5, wpk + OFF_WG1 + nt * 5 * 1024, lane, bg1[nt * 16 + mlo]);
#pragma unroll
      for (int g = 0; g < 8; ++g)
        bbuf[(g + 8 * hiq) * 256 + nt * 16 + mlo] = f2bf(fmaxf(acc[g], 0.0f));
    }
    __syncthreads();

    // ---- gate = sigmoid(g1@wg2+bg2) -> gbuf ----
    for (int nt = wid; nt < 8; nt += 4) {
      v8f acc = wgemm(bbuf, 0, 8, wpk + OFF_WG2 + nt * 8 * 1024, lane, bg2[nt * 16 + mlo]);
#pragma unroll
      for (int g = 0; g < 8; ++g)
        gbuf[(g + 8 * hiq) * 128 + nt * 16 + mlo] = sigmoidf_(acc[g]);
    }
    __syncthreads();

    // ---- p1 = relu(h@wp1+bp1) -> bbuf ----
    for (int nt = wid; nt < 16; nt += 4) {
      v8f acc = wgemm(abuf, 0, 5, wpk + OFF_WP1 + nt * 5 * 1024, lane, bp1[nt * 16 + mlo]);
#pragma unroll
      for (int g = 0; g < 8; ++g)
        bbuf[(g + 8 * hiq) * 256 + nt * 16 + mlo] = f2bf(fmaxf(acc[g], 0.0f));
    }
    __syncthreads();

    // ---- prop = p1@wp2+bp2 -> pbuf ----
    for (int nt = wid; nt < 8; nt += 4) {
      v8f acc = wgemm(bbuf, 0, 8, wpk + OFF_WP2 + nt * 8 * 1024, lane, bp2[nt * 16 + mlo]);
#pragma unroll
      for (int g = 0; g < 8; ++g)
        pbuf[(g + 8 * hiq) * 128 + nt * 16 + mlo] = acc[g];
    }
    __syncthreads();

    // ---- z_loc = (1-gate)*(z@wlin+blin) + gate*prop -> gbuf ----
    for (int nt = wid; nt < 8; nt += 4) {
      v8f acc = wgemm(abuf, 32, 4, wpk + OFF_WLIN + nt * 4 * 1024, lane, blin[nt * 16 + mlo]);
#pragma unroll
      for (int g = 0; g < 8; ++g) {
        int o = (g + 8 * hiq) * 128 + nt * 16 + mlo;
        float ga = gbuf[o];
        gbuf[o] = (1.0f - ga) * acc[g] + ga * pbuf[o];
      }
    }
    __syncthreads();

    // ---- relu(prop) -> abuf bf16 cols 0..127 ----
    for (int idx = tid; idx < 16 * 128; idx += 128) {
      int m = idx >> 7, n = idx & 127;
      abuf[m * 256 + n] = f2bf(fmaxf(pbuf[idx], 0.0f));
    }
    __syncthreads();

    // ---- z_scale = softplus(.@wsig+bsig); x0 = z_loc + s*eps; base_lp ----
    for (int nt = wid; nt < 8; nt += 4) {
      v8f acc = wgemm(abuf, 0, 4, wpk + OFF_WSIG + nt * 4 * 1024, lane, bsig[nt * 16 + mlo]);
#pragma unroll
      for (int g = 0; g < 8; ++g) {
        int m = g + 8 * hiq, n = nt * 16 + mlo, o = m * 128 + n;
        float s  = softplusf_(acc[g]);
        float e  = eps[(size_t)(t * 1024 + rbase + m) * 128 + n];
        float x0 = gbuf[o] + s * e;
        xbuf[o] = x0;
        bbuf[m * 256 + n] = f2bf(x0);
        lpacc[g] += -0.5f * e * e - __logf(s) - 0.91893853320467274f;  // 0.5*log(2pi)
      }
    }
    __syncthreads();

    // ---- 2x IAF: h=relu(z@wh+bh); out=h@wo+bo; y=x*exp(clip(ls))+m ----
    for (int fi = 0; fi < 2; ++fi) {
      const unsigned char* WH = wpk + (fi ? OFF_WH1 : OFF_WH0);
      const unsigned char* WO = wpk + (fi ? OFF_WO1 : OFF_WO0);
      const float* bh = fi ? bh1 : bh0;
      const float* bo = fi ? bo1 : bo0;

      for (int nt = wid; nt < 16; nt += 4) {
        v8f acc = wgemm(bbuf, 0, 4, WH + nt * 4 * 1024, lane, bh[nt * 16 + mlo]);
#pragma unroll
        for (int g = 0; g < 8; ++g)
          abuf[(g + 8 * hiq) * 256 + nt * 16 + mlo] = f2bf(fmaxf(acc[g], 0.0f));
      }
      __syncthreads();

      // nt<8 produce m (same wave later consumes at nt+8 -> same lane, in order)
      for (int nt = wid; nt < 16; nt += 4) {
        v8f acc = wgemm(abuf, 0, 8, WO + nt * 8 * 1024, lane, bo[nt * 16 + mlo]);
        if (nt < 8) {
#pragma unroll
          for (int g = 0; g < 8; ++g)
            mbuf[(g + 8 * hiq) * 128 + nt * 16 + mlo] = acc[g];
        } else {
#pragma unroll
          for (int g = 0; g < 8; ++g) {
            int m = g + 8 * hiq, n = (nt - 8) * 16 + mlo, o = m * 128 + n;
            float ls = fminf(fmaxf(acc[g], -5.0f), 3.0f);
            float y  = xbuf[o] * __expf(ls) + mbuf[o];
            xbuf[o] = y;
            bbuf[m * 256 + n] = f2bf(y);
            lpacc[g] -= ls;
          }
        }
      }
      __syncthreads();
    }

    // ---- emit z samples ----
    for (int idx = tid; idx < 16 * 128; idx += 128)
      outZ[(size_t)(t * 1024 + rbase) * 128 + idx] = xbuf[idx];

    // ---- energy head: e1, e2, e3 ----
    for (int nt = wid; nt < 16; nt += 4) {
      v8f acc = wgemm(bbuf, 0, 4, wpk + OFF_WE1 + nt * 4 * 1024, lane, be1[nt * 16 + mlo]);
#pragma unroll
      for (int g = 0; g < 8; ++g)
        abuf[(g + 8 * hiq) * 256 + nt * 16 + mlo] = f2bf(fmaxf(acc[g], 0.0f));
    }
    __syncthreads();
    for (int nt = wid; nt < 16; nt += 4) {
      v8f acc = wgemm(abuf, 0, 8, wpk + OFF_WE2 + nt * 8 * 1024, lane, be2[nt * 16 + mlo]);
#pragma unroll
      for (int g = 0; g < 8; ++g)
        bbuf[(g + 8 * hiq) * 256 + nt * 16 + mlo] = f2bf(fmaxf(acc[g], 0.0f));
    }
    __syncthreads();
    for (int nt = wid; nt < 2; nt += 4) {
      int n = nt * 16 + mlo;
      float bv = (n < 28) ? be3[n] : 0.0f;
      v8f acc = wgemm(bbuf, 0, 8, wpk + OFF_WE3 + nt * 8 * 1024, lane, bv);
#pragma unroll
      for (int g = 0; g < 8; ++g) {
        if (n < 28) {
          int m = g + 8 * hiq;
          size_t base = (size_t)(t * 1024 + rbase + m);
          int c = n >> 2, j = n & 3;
          float v = acc[g];
          if (j < 3) {
            float sg = sigmoidf_(v);
            float uu = u[base * 21 + c * 3 + j];
            float xs = (uu < sg) ? 1.0f : 0.0f;
            outXS[base * 21 + c * 3 + j]  = xs;
            outXLP[base * 21 + c * 3 + j] = xs * v - softplusf_(v);
          } else {
            outFR[base * 7 + c] = sigmoidf_(v);
          }
        }
      }
    }

    // ---- z_lp = base_lp - sum(log_s): reduce across 16 lanes, then waves ----
#pragma unroll
    for (int g = 0; g < 8; ++g) {
      float v = lpacc[g];
      v += __shfl_xor(v, 1, 16);
      v += __shfl_xor(v, 2, 16);
      v += __shfl_xor(v, 4, 16);
      v += __shfl_xor(v, 8, 16);
      if (mlo == 0) atomicAdd(&lpbuf[g + 8 * hiq], v);
    }
    __syncthreads();
    if (tid < 16) outZLP[t * 1024 + rbase + tid] = lpbuf[tid];
    __syncthreads();
  }
}

// ---------------------------------------------------------------------------
// Assumed d_in order: setup dict order (input, z0, eps, u) then params
// tree-flattened with sorted keys:
//  4 be1  5 be2  6 be3  7 bg1  8 bg2  9 blin 10 bp1 11 bp2 12 bsig
// 13 bh0 14 bo0 15 wh0 16 wo0 17 bh1 18 bo1 19 wh1 20 wo1
// 21 we1 22 we2 23 we3 24 wg1 25 wg2 26 wlin 27 wp1 28 wp2 29 wsig
// ---------------------------------------------------------------------------
extern "C" void kernel_launch(void* const* d_in, const int* in_sizes, int n_in,
                              void* d_out, int out_size, void* d_ws, size_t ws_size,
                              hipStream_t stream) {
  unsigned char* ws = (unsigned char*)d_ws;
  auto F = [&](int i) { return (const float*)d_in[i]; };

  pack_w<<<80,  32, 0, stream>>>(F(24), (unsigned int*)(ws + OFF_WG1), 160, 256, 5);
  pack_w<<<64,  32, 0, stream>>>(F(25), (unsigned int*)(ws + OFF_WG2), 256, 128, 8);
  pack_w<<<80,  32, 0, stream>>>(F(27), (unsigned int*)(ws + OFF_WP1), 160, 256, 5);
  pack_w<<<64,  32, 0, stream>>>(F(28), (unsigned int*)(ws + OFF_WP2), 256, 128, 8);
  pack_w<<<32,  32, 0, stream>>>(F(26), (unsigned int*)(ws + OFF_WLIN), 128, 128, 4);
  pack_w<<<32,  32, 0, stream>>>(F(29), (unsigned int*)(ws + OFF_WSIG), 128, 128, 4);
  pack_w<<<64,  32, 0, stream>>>(F(15), (unsigned int*)(ws + OFF_WH0), 128, 256, 4);
  pack_w<<<128, 32, 0, stream>>>(F(16), (unsigned int*)(ws + OFF_WO0), 256, 256, 8);
  pack_w<<<64,  32, 0, stream>>>(F(19), (unsigned int*)(ws + OFF_WH1), 128, 256, 4);
  pack_w<<<128, 32, 0, stream>>>(F(20), (unsigned int*)(ws + OFF_WO1), 256, 256, 8);
  pack_w<<<64,  32, 0, stream>>>(F(21), (unsigned int*)(ws + OFF_WE1), 128, 256, 4);
  pack_w<<<128, 32, 0, stream>>>(F(22), (unsigned int*)(ws + OFF_WE2), 256, 256, 8);
  pack_w<<<16,  32, 0, stream>>>(F(23), (unsigned int*)(ws + OFF_WE3), 256, 28, 8);

  ssm_scan<<<64, 128, 0, stream>>>(
      F(0), F(1), F(2), F(3),
      F(7), F(8), F(10), F(11), F(9), F(12),
      F(13), F(14), F(17), F(18),
      F(4), F(5), F(6),
      ws, (float*)d_out);
}
